// GlobalDiscreteAttention_8160437863061
// MI455X (gfx1250) — compile-verified, matrix operand order checked by reading, not run on hardware
//
#include <hip/hip_runtime.h>
#include <cstdint>

typedef __attribute__((ext_vector_type(16))) _Float16 v16h;
typedef __attribute__((ext_vector_type(8)))  float    v8f;

#define NPOS 4096   // H*W = 64*64
#define NBATCH 8
#define DV 32

// ---------------------------------------------------------------------------
// Kernel 1: bitpack z -> byte codes, and per-batch 256-bin histogram.
// grid (16, 8), block 256. Reads of z are coalesced across n.
// ---------------------------------------------------------------------------
__global__ void k_pack_hist(const float* __restrict__ z,
                            uint8_t* __restrict__ codes,
                            int* __restrict__ hist) {
  __shared__ int lh[256];
  const int tid = threadIdx.x;
  lh[tid] = 0;
  __syncthreads();

  const int b = blockIdx.y;
  const int n = blockIdx.x * 256 + tid;
  const float* zb = z + (size_t)b * 8 * NPOS + n;

  unsigned code = 0;
#pragma unroll
  for (int i = 0; i < 8; ++i) {
    code |= (zb[i * NPOS] > 0.5f ? 1u : 0u) << i;
  }
  codes[b * NPOS + n] = (uint8_t)code;
  atomicAdd(&lh[code], 1);
  __syncthreads();

  if (lh[tid] != 0) atomicAdd(&hist[b * 256 + tid], lh[tid]);
}

// ---------------------------------------------------------------------------
// Kernel 2: per-batch code-space attention as a 256x256 @ 256x48 GEMM with
// v_wmma_f32_16x16x32_f16.
//   A[a][c] = w^popc(a^c), w = exp(-1/T)   -- factored, built in registers:
//     popc(a^c) = popc(mn^(k&15)) + ((mt&1)^(k>>4)) + popc(wave^kb)
//     => A-frag = U[mt&1] * scalar(wave^kb)        (8 pk_mul per K-step)
//   B[c][0..31] = (cnt[c]/4096)*embed[c][:], B[c][32] = cnt[c]/4096, rest 0
//     -- stored in LDS pre-swizzled in WMMA fragment order (ds_load_b128 x2)
// res[b][a][j] = C[a][j] / C[a][32]
// grid 8, block 256 (8 waves). Each wave: 2 M-tiles x 3 N-tiles x 8 K-steps.
// ---------------------------------------------------------------------------
__global__ void k_code_gemm(const int* __restrict__ hist,
                            const float* __restrict__ value_embed,
                            const float* __restrict__ temperature,
                            float* __restrict__ res) {
  // Frag-linear B: [kb][nt][lane][16 halves]  (8*3*32*16 halves = 24 KB)
  __shared__ _Float16 Bfrag[8 * 3 * 32 * 16];
  __shared__ float csh[256];   // normalized per-code counts

  const int tid = threadIdx.x;
  const int b = blockIdx.x;

  const float T = fmaxf(temperature[0], 0.1f);
  const float w = __expf(-1.0f / T);

  csh[tid] = (float)hist[b * 256 + tid] * (1.0f / 4096.0f);
  __syncthreads();

  // ---- Build B fragments (one-time, 3 lane-frags per thread) --------------
#pragma unroll
  for (int r = 0; r < 3; ++r) {
    const int f = tid + r * 256;          // frag-lane id, 0..767
    const int lane_f = f & 31;
    const int nt = (f >> 5) % 3;
    const int kb = (f >> 5) / 3;
    const int h = lane_f >> 4;
    const int j = nt * 16 + (lane_f & 15);   // B column (0..47)
    _Float16* dst = &Bfrag[f * 16];
#pragma unroll
    for (int i = 0; i < 16; ++i) {
      const int k = i + (i >= 8 ? 8 : 0) + h * 8;   // 16-bit frag K mapping
      const int c = kb * 32 + k;
      float v;
      if (j < 32)       v = csh[c] * value_embed[c * 32 + j];
      else if (j == 32) v = csh[c];                  // denominator column
      else              v = 0.0f;                    // padding columns
      dst[i] = (_Float16)v;
    }
  }
  __syncthreads();

  const int lane = tid & 31;
  const int wave = tid >> 5;
  const int h  = lane >> 4;
  const int mn = lane & 15;

  // ---- Per-lane factored A rows (one-time, 8 VGPRs each) -----------------
  // U[p][i] = w^( popc(mn ^ (k&15)) + (p ^ (k>>4)) )
  v16h U0, U1;
#pragma unroll
  for (int i = 0; i < 16; ++i) {
    const int k = i + (i >= 8 ? 8 : 0) + h * 8;
    const int d1 = __popc(mn ^ (k & 15));
    const int k4 = (k >> 4) & 1;
    U0[i] = (_Float16)__expf(-(float)(d1 + (0 ^ k4)) / T);
    U1[i] = (_Float16)__expf(-(float)(d1 + (1 ^ k4)) / T);
  }

#pragma unroll
  for (int t = 0; t < 2; ++t) {
    const int mt = wave * 2 + t;          // M tile; mt&1==t, mt>>1==wave
    const v16h U = t ? U1 : U0;           // compile-time after unroll

    v8f acc0 = {};   // value columns 0..15
    v8f acc1 = {};   // value columns 16..31
    v8f acc2 = {};   // column 32 = denominator

#pragma unroll 2
    for (int kb = 0; kb < 8; ++kb) {
      // Wave-uniform scalar factor w^popc(wave ^ kb)
      const int x = wave ^ kb;
      const float s = ((x & 1) ? w : 1.0f) *
                      ((x & 2) ? w : 1.0f) *
                      ((x & 4) ? w : 1.0f);
      const _Float16 sh = (_Float16)s;

      v16h a;
#pragma unroll
      for (int i = 0; i < 16; ++i) a[i] = U[i] * sh;   // v_pk_mul_f16 x8

      const _Float16* bp = &Bfrag[(kb * 3) * (32 * 16) + lane * 16];
      const v16h b0 = *(const v16h*)(bp);
      const v16h b1 = *(const v16h*)(bp + 32 * 16);
      const v16h b2 = *(const v16h*)(bp + 2 * 32 * 16);

      acc0 = __builtin_amdgcn_wmma_f32_16x16x32_f16(false, a, false, b0,
                                                    (short)0, acc0, false, false);
      acc1 = __builtin_amdgcn_wmma_f32_16x16x32_f16(false, a, false, b1,
                                                    (short)0, acc1, false, false);
      acc2 = __builtin_amdgcn_wmma_f32_16x16x32_f16(false, a, false, b2,
                                                    (short)0, acc2, false, false);
    }

    // C layout: VGPR r -> row (r + 8h), lane -> col (lane&15).
    // Denominator (col 32) lives in lanes n==0 (lane 0 / lane 16); broadcast
    // to each half with a wave32 shuffle from lane (l & 16).
    float* rb = res + ((size_t)b * 256 + mt * 16) * DV;
#pragma unroll
    for (int r = 0; r < 8; ++r) {
      float den = __shfl(acc2[r], lane & 16, 32);
      den = fmaxf(den, 1e-30f);
      const float inv = 1.0f / den;
      const int arow = r + 8 * h;
      rb[arow * DV + mn]      = acc0[r] * inv;
      rb[arow * DV + 16 + mn] = acc1[r] * inv;
    }
  }
}

// ---------------------------------------------------------------------------
// Kernel 3: scatter per-code results to (B, 32, H, W). Fully coalesced writes.
// ---------------------------------------------------------------------------
__global__ void k_scatter(const uint8_t* __restrict__ codes,
                          const float* __restrict__ res,
                          float* __restrict__ out) {
  const int idx = blockIdx.x * 256 + threadIdx.x;   // over B*32*4096 = 1M
  const int b = idx >> 17;          // 32*4096 = 2^17
  const int j = (idx >> 12) & 31;
  const int n = idx & (NPOS - 1);
  const unsigned code = codes[b * NPOS + n];
  out[idx] = res[((b * 256) + code) * DV + j];
}

// ---------------------------------------------------------------------------
extern "C" void kernel_launch(void* const* d_in, const int* in_sizes, int n_in,
                              void* d_out, int out_size, void* d_ws, size_t ws_size,
                              hipStream_t stream) {
  const float* z           = (const float*)d_in[0];   // (8, 8, 64, 64)
  const float* temperature = (const float*)d_in[1];   // scalar
  const float* value_embed = (const float*)d_in[2];   // (256, 32)
  float* out = (float*)d_out;                         // (8, 32, 64, 64)

  uint8_t* codes = (uint8_t*)d_ws;                           // 32768 B
  int*     hist  = (int*)((char*)d_ws + 32768);              //  8192 B
  float*   res   = (float*)((char*)d_ws + 32768 + 8192);     // 262144 B

  hipMemsetAsync(hist, 0, NBATCH * 256 * sizeof(int), stream);

  dim3 g1(NPOS / 256, NBATCH);
  k_pack_hist<<<g1, 256, 0, stream>>>(z, codes, hist);

  k_code_gemm<<<NBATCH, 256, 0, stream>>>(hist, value_embed, temperature, res);

  k_scatter<<<(NBATCH * DV * NPOS) / 256, 256, 0, stream>>>(codes, res, out);
}
